// HGA_53987738911523
// MI455X (gfx1250) — matrix operand order repaired and checked by reading, not compiled, not run on hardware
//
#include <hip/hip_runtime.h>
#include <hip/hip_bf16.h>
#include <math.h>

// ---------------------------------------------------------------------------
// HGA pipeline for MI455X (gfx1250, wave32).
// Memory-bound: two streaming passes over x (629 MB each) + 105 MB out
// => ~1.37 GB / 23.3 TB/s ~ 59 us floor. Streaming kernels use b64/b128
// vector memory ops to maximize bytes per vmem instruction.
// Middle GEMM (64x256 @ 256x9600) uses v_wmma_f32_16x16x32_f16.
// ---------------------------------------------------------------------------

#define NN   32
#define CC   256
#define LL   6
#define TT   64
#define VV   50
#define CI   64
#define KNB  3
#define COLS (NN * LL * VV)   // 9600
#define LV   (LL * VV)        // 300
#define TV   (TT * VV)        // 3200
#define V2   (VV / 2)         // 25 float2 per row
#define J2   (NN * LL * V2)   // 4800 float2-columns per channel
#define EPSV 1e-5f

// LAYERS baked from the reference (compile-time constants)
__constant__ int LAY_OFF[7] = {0, 6, 20, 42, 62, 78, 94};
__constant__ int LAY_LEN[6] = {6, 14, 22, 20, 16, 16};
__constant__ int LAY_IDX[94] = {
    1, 0, 20, 26, 25, 45,
    0, 20, 12, 16, 2, 4, 8, 25, 45, 37, 41, 27, 29, 33,
    12, 16, 2, 4, 8, 13, 17, 3, 5, 9, 3, 28, 37, 41, 27, 29, 33, 38, 42, 28, 30, 34,
    13, 17, 3, 5, 9, 14, 18, 6, 10, 3, 28, 38, 42, 28, 30, 34, 39, 43, 31, 35,
    14, 18, 6, 10, 15, 19, 7, 11, 39, 43, 31, 35, 40, 44, 32, 36,
    15, 19, 7, 11, 21, 22, 23, 24, 40, 44, 32, 36, 46, 47, 48, 49};

typedef __attribute__((ext_vector_type(16))) _Float16 v16h;
typedef __attribute__((ext_vector_type(8)))  float    v8f;
typedef __attribute__((ext_vector_type(2)))  float    f32x2;
typedef __attribute__((ext_vector_type(4)))  float    f32x4;

// Workspace layout (float offsets)
#define WS_XT     0u                       // (C, COLS)        2,457,600
#define WS_Y      2457600u                 // (CI, COLS)         614,400
#define WS_MEAN1  3072000u                 // 64
#define WS_RSQ1   3072064u                 // 64
#define WS_XS     3072128u                 // (N, CI, L)          12,288
#define WS_EPRE   3084416u                 // (N, CI, 18)         36,864
#define WS_MEAN2  3121280u                 // 64
#define WS_RSQ2   3121344u                 // 64
#define WS_EMAX   3121408u                 // (N, CI, L)          12,288
#define WS_ATT    3133696u                 // (N, C, L)           49,152
#define WS_TOTAL  3182848u

// ---------------------------------------------------------------------------
// Pass 1: x_t[c, j] = max_t x[n,c,l,t,v].  float2 lanes (global_load_b64),
// 4 independent max accumulators so the 64 strided row loads pipeline.
// Streams 629 MB, fully coalesced along v.
// ---------------------------------------------------------------------------
__global__ void k_tmax(const float* __restrict__ x, float* __restrict__ ws) {
    unsigned g  = blockIdx.x * 256u + threadIdx.x;  // 1,228,800 threads
    unsigned c  = g / J2;
    unsigned j2 = g % J2;
    unsigned n  = j2 / (LL * V2);
    unsigned r  = j2 % (LL * V2);
    unsigned l  = r / V2;
    unsigned v2 = r % V2;
    const f32x2* p = (const f32x2*)(x + ((((size_t)n * CC + c) * LL + l) * TT) * VV + 2 * v2);
    // stride between t-rows is VV floats = 25 float2
    f32x2 m0 = p[0 * V2], m1 = p[1 * V2], m2 = p[2 * V2], m3 = p[3 * V2];
#pragma unroll 4
    for (int t = 4; t < TT; t += 4) {
        f32x2 a = p[(size_t)(t + 0) * V2];
        f32x2 b = p[(size_t)(t + 1) * V2];
        f32x2 cc2 = p[(size_t)(t + 2) * V2];
        f32x2 d = p[(size_t)(t + 3) * V2];
        m0.x = fmaxf(m0.x, a.x);  m0.y = fmaxf(m0.y, a.y);
        m1.x = fmaxf(m1.x, b.x);  m1.y = fmaxf(m1.y, b.y);
        m2.x = fmaxf(m2.x, cc2.x); m2.y = fmaxf(m2.y, cc2.y);
        m3.x = fmaxf(m3.x, d.x);  m3.y = fmaxf(m3.y, d.y);
    }
    f32x2 m;
    m.x = fmaxf(fmaxf(m0.x, m1.x), fmaxf(m2.x, m3.x));
    m.y = fmaxf(fmaxf(m0.y, m1.y), fmaxf(m2.y, m3.y));
    f32x2* o = (f32x2*)(ws + WS_XT + (size_t)c * COLS + n * LV + l * VV + 2 * v2);
    *o = m;
}

// ---------------------------------------------------------------------------
// Down-projection GEMM with WMMA:  Y(64 x 9600) = W_down(64x256) * XT(256x9600)
// Block = 128 threads (4 waves); wave w owns M-tile d0 = 16*w; grid = 600
// column tiles. K = 256 in 8 chunks of 32, f16 inputs / f32 accum.
// A-frag (16-bit A 16x32): lanes 0-15 -> M=lane, VGPR r holds K={2r,2r+1}
// for r<4 else {16+2(r-4),...}; lanes 16-31 same M, K+8.
// B-frag (16-bit B 32x16): lanes 0-15 -> N=lane, elements K=0..7,16..23;
// lanes 16-31 -> N=lane-16, elements K=8..15,24..31.
// C/D: lanes 0-15 -> N=lane, VGPR r -> M=r; lanes 16-31 -> M=8+r.
// ---------------------------------------------------------------------------
__global__ void k_gemm_down(const float* __restrict__ Wd,
                            const float* __restrict__ bdown,
                            float* __restrict__ ws) {
    const float* xt = ws + WS_XT;
    float*       y  = ws + WS_Y;

    int lane = threadIdx.x & 31;
    int d0   = (threadIdx.x >> 5) * 16;     // 0,16,32,48
    int j0   = blockIdx.x * 16;             // 600 tiles

    int half = lane >> 4;
    int mrow = lane & 15;

    v8f acc = {};
    for (int kc = 0; kc < 8; ++kc) {
        int c0 = kc * 32;
        v16h af, bf;
#pragma unroll
        for (int r = 0; r < 8; ++r) {
            int kbase = ((r < 4) ? (2 * r) : (16 + 2 * (r - 4))) + half * 8;
            af[2 * r]     = (_Float16)Wd[(d0 + mrow) * CC + c0 + kbase];
            af[2 * r + 1] = (_Float16)Wd[(d0 + mrow) * CC + c0 + kbase + 1];
        }
#pragma unroll
        for (int e = 0; e < 16; ++e) {
            int kk = ((e < 8) ? e : (8 + e)) + half * 8;
            bf[e] = (_Float16)xt[(size_t)(c0 + kk) * COLS + j0 + mrow];
        }
        acc = __builtin_amdgcn_wmma_f32_16x16x32_f16(
            false, af, false, bf, (short)0, acc, false, false);
    }

    int colN  = lane & 15;
    int mBase = half * 8;
#pragma unroll
    for (int r = 0; r < 8; ++r) {
        int d = d0 + mBase + r;
        y[(size_t)d * COLS + j0 + colN] = acc[r] + bdown[d];
    }
}

// ---------------------------------------------------------------------------
// BN1 stats: per-channel mean / rsqrt(var+eps) over 9600 values.
// grid = 64 (one block per d), 256 threads.
// ---------------------------------------------------------------------------
__global__ void k_stats1(float* __restrict__ ws) {
    __shared__ float ss[256], sq[256];
    int d = blockIdx.x, t = threadIdx.x;
    const float* y = ws + WS_Y + (size_t)d * COLS;
    float s = 0.f, q = 0.f;
    for (int i = t; i < COLS; i += 256) { float v = y[i]; s += v; q += v * v; }
    ss[t] = s; sq[t] = q;
    __syncthreads();
    for (int o = 128; o > 0; o >>= 1) {
        if (t < o) { ss[t] += ss[t + o]; sq[t] += sq[t + o]; }
        __syncthreads();
    }
    if (t == 0) {
        float mean = ss[0] / (float)COLS;
        float var  = sq[0] / (float)COLS - mean * mean;
        ws[WS_MEAN1 + d] = mean;
        ws[WS_RSQ1 + d]  = rsqrtf(var + EPSV);
    }
}

// ---------------------------------------------------------------------------
// xs[n,d,l] = mean over LAYERS[l] of relu(BN1(y)).  grid = N*L, block = 64 (d)
// ---------------------------------------------------------------------------
__global__ void k_xs(const float* __restrict__ g1, const float* __restrict__ b1,
                     float* __restrict__ ws) {
    int n = blockIdx.x / LL, l = blockIdx.x % LL, d = threadIdx.x;
    float mean = ws[WS_MEAN1 + d], rsq = ws[WS_RSQ1 + d];
    float ga = g1[d], be = b1[d];
    int len = LAY_LEN[l], off = LAY_OFF[l];
    const float* y = ws + WS_Y + (size_t)d * COLS + n * LV + l * VV;
    float acc = 0.f;
    for (int i = 0; i < len; ++i) {
        float v = (y[LAY_IDX[off + i]] - mean) * rsq * ga + be;
        acc += fmaxf(v, 0.f);
    }
    ws[WS_XS + (n * CI + d) * LL + l] = acc / (float)len;
}

// ---------------------------------------------------------------------------
// Per-sample graph: pairwise dist, top-k (jax tie-break: lowest index wins),
// edge features, edge GEMM pre-activation e_pre (64 x 18).
// grid = N, block = 64.
// ---------------------------------------------------------------------------
__global__ void k_graph(const float* __restrict__ We, float* __restrict__ ws) {
    __shared__ float xs_s[CI][LL];
    __shared__ float inner_s[LL][LL];
    __shared__ int   idx_s[LL][KNB];
    int n = blockIdx.x, t = threadIdx.x;

    for (int l = 0; l < LL; ++l)
        xs_s[t][l] = ws[WS_XS + (n * CI + t) * LL + l];
    __syncthreads();

    if (t < LL * LL) {
        int l = t / LL, m = t % LL;
        float s = 0.f;
        for (int d = 0; d < CI; ++d) s += xs_s[d][l] * xs_s[d][m];
        inner_s[l][m] = s;
    }
    __syncthreads();

    if (t < LL) {
        int l = t;
        float pd[LL];
        bool taken[LL];
        for (int m = 0; m < LL; ++m) {
            pd[m] = 2.f * inner_s[l][m] - inner_s[l][l] - inner_s[m][m];
            taken[m] = false;
        }
        for (int k = 0; k < KNB; ++k) {
            float best = -INFINITY; int bi = 0;
            for (int m = 0; m < LL; ++m)
                if (!taken[m] && pd[m] > best) { best = pd[m]; bi = m; }
            taken[bi] = true;
            idx_s[l][k] = bi;
        }
    }
    __syncthreads();

    // e_pre[d, l, k] = sum_c We[d,c]*(xs[c,idx]-xs[c,l]) + We[d,64+c]*xs[c,l]
    int d = t;
    for (int lk = 0; lk < LL * KNB; ++lk) {
        int l = lk / KNB, k = lk % KNB, jn = idx_s[l][k];
        float acc = 0.f;
        const float* w = We + d * (2 * CI);
        for (int c = 0; c < CI; ++c) {
            acc += w[c] * (xs_s[c][jn] - xs_s[c][l]);
            acc += w[CI + c] * xs_s[c][l];
        }
        ws[WS_EPRE + (n * CI + d) * (LL * KNB) + lk] = acc;
    }
}

// ---------------------------------------------------------------------------
// BN2 stats over (n,l,k) = 576 values per d.  grid = 64, block = 64.
// ---------------------------------------------------------------------------
__global__ void k_stats2(float* __restrict__ ws) {
    __shared__ float ss[64], sq[64];
    int d = blockIdx.x, t = threadIdx.x;
    float s = 0.f, q = 0.f;
    for (int i = t; i < NN * LL * KNB; i += 64) {
        int n = i / (LL * KNB), lk = i % (LL * KNB);
        float v = ws[WS_EPRE + (n * CI + d) * (LL * KNB) + lk];
        s += v; q += v * v;
    }
    ss[t] = s; sq[t] = q;
    __syncthreads();
    for (int o = 32; o > 0; o >>= 1) {
        if (t < o) { ss[t] += ss[t + o]; sq[t] += sq[t + o]; }
        __syncthreads();
    }
    if (t == 0) {
        float cnt  = (float)(NN * LL * KNB);
        float mean = ss[0] / cnt;
        float var  = sq[0] / cnt - mean * mean;
        ws[WS_MEAN2 + d] = mean;
        ws[WS_RSQ2 + d]  = rsqrtf(var + EPSV);
    }
}

// ---------------------------------------------------------------------------
// emax[n,d,l] = max_k leakyrelu(BN2(e_pre), 0.2).  12288 threads.
// ---------------------------------------------------------------------------
__global__ void k_emax(const float* __restrict__ g2, const float* __restrict__ b2,
                       float* __restrict__ ws) {
    int g = blockIdx.x * 256 + threadIdx.x;
    if (g >= NN * CI * LL) return;
    int n = g / (CI * LL), r = g % (CI * LL), d = r / LL, l = r % LL;
    float mean = ws[WS_MEAN2 + d], rsq = ws[WS_RSQ2 + d];
    float ga = g2[d], be = b2[d];
    float m = -INFINITY;
    for (int k = 0; k < KNB; ++k) {
        float v = ws[WS_EPRE + (n * CI + d) * (LL * KNB) + l * KNB + k];
        v = (v - mean) * rsq * ga + be;
        v = (v >= 0.f) ? v : 0.2f * v;
        m = fmaxf(m, v);
    }
    ws[WS_EMAX + (n * CI + d) * LL + l] = m;
}

// ---------------------------------------------------------------------------
// att[n,c,l] = sigmoid(b_agg[c] + sum_d W_agg[c,d]*emax[n,d,l]).
// grid = N*L, block = 256 (c).
// ---------------------------------------------------------------------------
__global__ void k_att(const float* __restrict__ Wa, const float* __restrict__ ba,
                      float* __restrict__ ws) {
    __shared__ float es[CI];
    int n = blockIdx.x / LL, l = blockIdx.x % LL, c = threadIdx.x;
    if (c < CI) es[c] = ws[WS_EMAX + (n * CI + c) * LL + l];
    __syncthreads();
    float acc = ba[c];
    const float* w = Wa + c * CI;
#pragma unroll 8
    for (int d = 0; d < CI; ++d) acc += w[d] * es[d];
    ws[WS_ATT + (n * CC + c) * LL + l] = 1.f / (1.f + expf(-acc));
}

// ---------------------------------------------------------------------------
// Pass 2: out[n,c,t,v] = sum_l x[n,c,l,t,v] * att[n,c,l].
// grid = N*C blocks, 256 threads sweep 800 float4 (global_load_b128 x6 +
// global_store_b128). Streams 629 MB in + 105 MB out.
// ---------------------------------------------------------------------------
__global__ void k_out(const float* __restrict__ x, const float* __restrict__ ws,
                      float* __restrict__ out) {
    int nc = blockIdx.x;                    // n*C + c
    __shared__ float a[LL];
    if (threadIdx.x < LL) a[threadIdx.x] = ws[WS_ATT + nc * LL + threadIdx.x];
    __syncthreads();
    float a0 = a[0], a1 = a[1], a2 = a[2], a3 = a[3], a4 = a[4], a5 = a[5];
    const f32x4* xb = (const f32x4*)(x + (size_t)nc * LL * TV);
    f32x4* ob = (f32x4*)(out + (size_t)nc * TV);
    const int Q = TV / 4;                   // 800 float4 per stream
    for (int i = threadIdx.x; i < Q; i += 256) {
        f32x4 s0 = xb[i];
        f32x4 s1 = xb[i + 1 * Q];
        f32x4 s2 = xb[i + 2 * Q];
        f32x4 s3 = xb[i + 3 * Q];
        f32x4 s4 = xb[i + 4 * Q];
        f32x4 s5 = xb[i + 5 * Q];
        f32x4 s;
        s.x = s0.x * a0 + s1.x * a1 + s2.x * a2 + s3.x * a3 + s4.x * a4 + s5.x * a5;
        s.y = s0.y * a0 + s1.y * a1 + s2.y * a2 + s3.y * a3 + s4.y * a4 + s5.y * a5;
        s.z = s0.z * a0 + s1.z * a1 + s2.z * a2 + s3.z * a3 + s4.z * a4 + s5.z * a5;
        s.w = s0.w * a0 + s1.w * a1 + s2.w * a2 + s3.w * a3 + s4.w * a4 + s5.w * a5;
        ob[i] = s;
    }
}

// ---------------------------------------------------------------------------
extern "C" void kernel_launch(void* const* d_in, const int* in_sizes, int n_in,
                              void* d_out, int out_size, void* d_ws, size_t ws_size,
                              hipStream_t stream) {
    (void)in_sizes; (void)n_in; (void)out_size;
    if (ws_size < (size_t)WS_TOTAL * sizeof(float)) return;

    const float* x      = (const float*)d_in[0];
    const float* W_down = (const float*)d_in[1];
    const float* b_down = (const float*)d_in[2];
    const float* gamma1 = (const float*)d_in[3];
    const float* beta1  = (const float*)d_in[4];
    const float* W_edge = (const float*)d_in[5];
    const float* gamma2 = (const float*)d_in[6];
    const float* beta2  = (const float*)d_in[7];
    const float* W_agg  = (const float*)d_in[8];
    const float* b_agg  = (const float*)d_in[9];
    float* out = (float*)d_out;
    float* ws  = (float*)d_ws;

    k_tmax     <<<(CC * J2) / 256, 256, 0, stream>>>(x, ws);
    k_gemm_down<<<COLS / 16, 128, 0, stream>>>(W_down, b_down, ws);
    k_stats1   <<<CI, 256, 0, stream>>>(ws);
    k_xs       <<<NN * LL, CI, 0, stream>>>(gamma1, beta1, ws);
    k_graph    <<<NN, CI, 0, stream>>>(W_edge, ws);
    k_stats2   <<<CI, 64, 0, stream>>>(ws);
    k_emax     <<<(NN * CI * LL + 255) / 256, 256, 0, stream>>>(gamma2, beta2, ws);
    k_att      <<<NN * LL, CC, 0, stream>>>(W_agg, b_agg, ws);
    k_out      <<<NN * CC, 256, 0, stream>>>(x, ws, out);
}
